// DeltaNet_31877247271430
// MI455X (gfx1250) — compile-verified, matrix operand order checked by reading, not run on hardware
//
#include <hip/hip_runtime.h>
#include <hip/hip_bf16.h>

typedef __bf16 bf16;
typedef bf16  v16bf __attribute__((ext_vector_type(16)));
typedef float v8f   __attribute__((ext_vector_type(8)));

union BFrag { v16bf v; bf16 e[16]; };
union FFrag { v8f  v; float e[8]; };

#define DKD   128
#define CSZ   32
#define LSEQ  4096
#define NCH   128
#define NBH   16

// padded leading dims (rows stay 16B aligned: 136*2=272=16*17, 40*2=80=16*5)
#define LDP   136   // 128-wide buffers
#define LDC   40    // 32-wide buffers

static __device__ __forceinline__ v8f wmma_bf(v16bf a, v16bf b, v8f c) {
  return __builtin_amdgcn_wmma_f32_16x16x32_bf16(false, a, false, b, (short)0, c, false, false);
}

static __device__ __forceinline__ v8f zero8() {
  v8f z = {0.f,0.f,0.f,0.f,0.f,0.f,0.f,0.f};
  return z;
}

// 16-bit fragment loader per CDNA5 ISA layout.
//   lane 0..15 -> idx = lane, K = {0..7}(v0..3) and {16..23}(v4..7)
//   lane 16..31 -> idx = lane-16, K = {8..15} and {24..31}
// idx_row=true  : element = buf[idx*ld + k]  -> two contiguous 16B groups (merges to ds_load_b128)
// idx_row=false : element = buf[k*ld + idx]  -> strided (use sparingly / hoisted)
static __device__ __forceinline__ v16bf ldfrag(const bf16* buf, int ld, int idx0, int k0,
                                               bool idx_row, int lane) {
  int idx = idx0 + (lane & 15);
  int hi  = (lane >> 4) & 1;
  BFrag f;
#pragma unroll
  for (int e = 0; e < 16; ++e) {
    int r = e >> 1, p = e & 1;
    int kk = k0 + 16 * (r >> 2) + 8 * hi + 2 * (r & 3) + p;
    f.e[e] = idx_row ? buf[idx * ld + kk] : buf[kk * ld + idx];
  }
  return f.v;
}

// ---------------------------------------------------------------------------
// Kernel 1: per-chunk precompute. One wave per chunk.
// ---------------------------------------------------------------------------
__global__ __launch_bounds__(32)
void delta_prep(const float* __restrict__ q, const float* __restrict__ k,
                const float* __restrict__ v, const float* __restrict__ beta,
                bf16* __restrict__ gq, bf16* __restrict__ gk,
                bf16* __restrict__ gw, float* __restrict__ gu) {
  __shared__ bf16 s_q [CSZ * LDP];
  __shared__ bf16 s_k [CSZ * LDP];
  __shared__ bf16 s_kb[CSZ * LDP];
  __shared__ bf16 s_v [CSZ * LDP];
  __shared__ float s_T[CSZ * 33];
  __shared__ bf16 s_att[CSZ * LDC];

  const int chunk = blockIdx.x;
  const int bh    = chunk >> 7;
  const int ch    = chunk & 127;
  const int lane  = threadIdx.x;
  const int row0  = bh * LSEQ + ch * CSZ;

  // ---- phase A: l2norm + beta scaling (lane = chunk row) ----
  {
    const int m = lane;
    const size_t g = (size_t)(row0 + m) * DKD;
    const float bm = beta[row0 + m];

    float ss = 1e-6f;
    for (int c = 0; c < DKD; ++c) { float x = q[g + c]; ss += x * x; }
    float rs = rsqrtf(ss);
    for (int c = 0; c < DKD; ++c) {
      bf16 bv = (bf16)(q[g + c] * rs);
      s_q[m * LDP + c] = bv;
      gq[g + c] = bv;
    }

    ss = 1e-6f;
    for (int c = 0; c < DKD; ++c) { float x = k[g + c]; ss += x * x; }
    rs = rsqrtf(ss);
    for (int c = 0; c < DKD; ++c) {
      float kn = k[g + c] * rs;
      bf16 bv = (bf16)kn;
      s_k[m * LDP + c]  = bv;
      s_kb[m * LDP + c] = (bf16)(kn * bm);
      gk[g + c] = bv;
    }

    for (int c = 0; c < DKD; ++c)
      s_v[m * LDP + c] = (bf16)(v[g + c] * bm);
  }
  __syncthreads();

  // ---- phase B: T = -(k_beta @ k^T), strictly lower ----
  for (int mt = 0; mt < 2; ++mt)
    for (int nt = 0; nt < 2; ++nt) {
      v8f acc = zero8();
#pragma unroll
      for (int kt = 0; kt < 4; ++kt) {
        v16bf a = ldfrag(s_kb, LDP, mt * 16, kt * 32, true, lane);
        v16bf b = ldfrag(s_k,  LDP, nt * 16, kt * 32, true, lane);
        acc = wmma_bf(a, b, acc);
      }
      FFrag d; d.v = acc;
      const int nn = nt * 16 + (lane & 15);
      const int hi = (lane >> 4) & 1;
#pragma unroll
      for (int r = 0; r < 8; ++r) {
        int mm = mt * 16 + r + 8 * hi;
        s_T[mm * 33 + nn] = (mm > nn) ? -d.e[r] : 0.f;
      }
    }
  __syncthreads();

  // ---- phase C: forward substitution (lane = column) ----
  {
    const int col = lane;
    for (int i = 1; i < CSZ; ++i) {
      __syncthreads();
      float rowv = s_T[i * 33 + col];
      float s = 0.f;
      for (int j = 0; j < i; ++j)
        s += __shfl(rowv, j, 32) * s_T[j * 33 + col];
      if (col < i) s_T[i * 33 + col] = rowv + s;
    }
    __syncthreads();
    s_T[col * 33 + col] = 1.f;
    __syncthreads();
    for (int i = 0; i < CSZ; ++i)
      s_att[i * LDC + col] = (bf16)s_T[i * 33 + col];
  }
  __syncthreads();

  // ---- phase D: u = att @ v, w = att @ k_beta (A-frags hoisted) ----
  {
    const int nn0 = lane & 15;
    const int hi  = (lane >> 4) & 1;
    v16bf a0 = ldfrag(s_att, LDC, 0,  0, true, lane);
    v16bf a1 = ldfrag(s_att, LDC, 16, 0, true, lane);
    for (int nt = 0; nt < 8; ++nt) {
      const int nn = nt * 16 + nn0;
      v16bf bv = ldfrag(s_v,  LDP, nt * 16, 0, false, lane);
      v16bf bk = ldfrag(s_kb, LDP, nt * 16, 0, false, lane);

      FFrag d0; d0.v = wmma_bf(a0, bv, zero8());
      FFrag d1; d1.v = wmma_bf(a1, bv, zero8());
#pragma unroll
      for (int r = 0; r < 8; ++r) {
        gu[(size_t)(row0 + r + 8 * hi) * DKD + nn]      = d0.e[r];
        gu[(size_t)(row0 + 16 + r + 8 * hi) * DKD + nn] = d1.e[r];
      }

      d0.v = wmma_bf(a0, bk, zero8());
      d1.v = wmma_bf(a1, bk, zero8());
#pragma unroll
      for (int r = 0; r < 8; ++r) {
        gw[(size_t)(row0 + r + 8 * hi) * DKD + nn]      = (bf16)d0.e[r];
        gw[(size_t)(row0 + 16 + r + 8 * hi) * DKD + nn] = (bf16)d1.e[r];
      }
    }
  }
}

// ---------------------------------------------------------------------------
// Kernel 2: sequential chunk scan. One workgroup (8 wave32) per (b,h).
// S kept f32 row-major + bf16 TRANSPOSED mirror ST[n][k] so all B-fragment
// loads are contiguous ds_load_b128. u_new kept only transposed (B-only use).
// ---------------------------------------------------------------------------
__global__ __launch_bounds__(256)
void delta_scan(const bf16* __restrict__ gq, const bf16* __restrict__ gk,
                const bf16* __restrict__ gw, const float* __restrict__ gu,
                float* __restrict__ o_out, float* __restrict__ S_out) {
  extern __shared__ char smem[];
  float* S     = (float*)smem;                 // 128x128 f32, ld 128 (64 KB)
  bf16*  ST    = (bf16*)(S + 128 * 128);       // S^T bf16: ST[n*LDP+m] (34 KB)
  bf16*  sunbT = ST + 128 * LDP;               // u_new^T: [n*LDC+k]    (10 KB)
  bf16*  sq    = sunbT + 128 * LDC;            // 32xLDP
  bf16*  sk    = sq + CSZ * LDP;
  bf16*  sw    = sk + CSZ * LDP;
  bf16*  satt  = sw + CSZ * LDP;               // 32xLDC

  const int bh   = blockIdx.x;
  const int t    = threadIdx.x;
  const int wid  = t >> 5;
  const int lane = t & 31;
  const int nn0  = lane & 15;
  const int hi   = (lane >> 4) & 1;

  for (int idx = t; idx < 128 * 128; idx += 256) S[idx] = 0.f;
  for (int idx = t; idx < 128 * LDP; idx += 256) ST[idx] = (bf16)0.f;
  __syncthreads();

  for (int ch = 0; ch < NCH; ++ch) {
    const size_t base = ((size_t)bh * LSEQ + (size_t)ch * CSZ) * DKD;

    // stage current chunk (padded rows)
    for (int idx = t; idx < CSZ * DKD; idx += 256) {
      const int row = idx >> 7, col = idx & 127;
      sq[row * LDP + col] = gq[base + idx];
      sk[row * LDP + col] = gk[base + idx];
      sw[row * LDP + col] = gw[base + idx];
    }
    // prefetch next chunk while this one computes (global_prefetch_b8)
    if (ch + 1 < NCH) {
      const size_t nb = base + (size_t)CSZ * DKD;
      if (t < 64) {
        __builtin_prefetch(gq + nb + (size_t)t * 64, 0, 1);
        __builtin_prefetch(gk + nb + (size_t)t * 64, 0, 1);
        __builtin_prefetch(gw + nb + (size_t)t * 64, 0, 1);
      } else if (t < 192) {
        __builtin_prefetch(gu + nb + (size_t)(t - 64) * 32, 0, 1);
      }
    }
    __syncthreads();

    // S^T B-fragments for this wave's column strip (nt = wid); reused in
    // steps 2 and 4 (S not mutated until step 5).
    v16bf BS[4];
#pragma unroll
    for (int kt = 0; kt < 4; ++kt)
      BS[kt] = ldfrag(ST, LDP, wid * 16, kt * 32, true, lane);

    // step 2: u_new = u - w @ S   (wave handles mt=0,1 at nt=wid)
    {
      const int nn = wid * 16 + nn0;
#pragma unroll
      for (int mt = 0; mt < 2; ++mt) {
        v8f acc = zero8();
#pragma unroll
        for (int kt = 0; kt < 4; ++kt) {
          v16bf a = ldfrag(sw, LDP, mt * 16, kt * 32, true, lane);
          acc = wmma_bf(a, BS[kt], acc);
        }
        FFrag d; d.v = acc;
#pragma unroll
        for (int r = 0; r < 8; ++r) {
          int mm = mt * 16 + r + 8 * hi;
          float val = gu[base + (size_t)mm * DKD + nn] - d.e[r];
          sunbT[nn * LDC + mm] = (bf16)val;   // transposed, r-contiguous
        }
      }
    }
    __syncthreads();

    // step 3: att_local = tril(q @ k^T)  (waves 0..3)
    if (wid < 4) {
      const int mt = wid >> 1, nt = wid & 1;
      v8f acc = zero8();
#pragma unroll
      for (int kt = 0; kt < 4; ++kt) {
        v16bf a = ldfrag(sq, LDP, mt * 16, kt * 32, true, lane);
        v16bf b = ldfrag(sk, LDP, nt * 16, kt * 32, true, lane);  // B = k^T
        acc = wmma_bf(a, b, acc);
      }
      FFrag d; d.v = acc;
      const int nn = nt * 16 + nn0;
#pragma unroll
      for (int r = 0; r < 8; ++r) {
        int mm = mt * 16 + r + 8 * hi;
        satt[mm * LDC + nn] = (mm >= nn) ? (bf16)d.e[r] : (bf16)0.f;
      }
    }
    __syncthreads();

    // step 4: o = q @ S + att_local @ u_new  (nt = wid, reuse BS)
    {
      const int nn = wid * 16 + nn0;
      v16bf bu = ldfrag(sunbT, LDC, wid * 16, 0, true, lane);
#pragma unroll
      for (int mt = 0; mt < 2; ++mt) {
        v8f acc = zero8();
#pragma unroll
        for (int kt = 0; kt < 4; ++kt) {
          v16bf a = ldfrag(sq, LDP, mt * 16, kt * 32, true, lane);
          acc = wmma_bf(a, BS[kt], acc);
        }
        v16bf a = ldfrag(satt, LDC, mt * 16, 0, true, lane);
        acc = wmma_bf(a, bu, acc);
        FFrag d; d.v = acc;
#pragma unroll
        for (int r = 0; r < 8; ++r) {
          int mm = mt * 16 + r + 8 * hi;
          o_out[base + (size_t)mm * DKD + nn] = d.e[r];
        }
      }
    }
    __syncthreads();

    // step 5: S += k^T @ u_new  (wave handles mt = wid over dk; hoisted A)
    {
      v16bf ak = ldfrag(sk, LDP, wid * 16, 0, false, lane);  // A = k^T (strided, 1/wave)
#pragma unroll
      for (int nt = 0; nt < 8; ++nt) {
        const int nn = nt * 16 + nn0;
        v16bf b = ldfrag(sunbT, LDC, nt * 16, 0, true, lane);
        FFrag c;
#pragma unroll
        for (int r = 0; r < 8; ++r)
          c.e[r] = S[(wid * 16 + r + 8 * hi) * DKD + nn];
        FFrag d; d.v = wmma_bf(ak, b, c.v);
#pragma unroll
        for (int r = 0; r < 8; ++r) {
          int mm = wid * 16 + r + 8 * hi;
          S[mm * DKD + nn]   = d.e[r];
          ST[nn * LDP + mm]  = (bf16)d.e[r];   // transposed, r-contiguous
        }
      }
    }
    __syncthreads();
  }

  for (int idx = t; idx < 128 * 128; idx += 256)
    S_out[(size_t)bh * 128 * 128 + idx] = S[idx];
}

extern "C" void kernel_launch(void* const* d_in, const int* in_sizes, int n_in,
                              void* d_out, int out_size, void* d_ws, size_t ws_size,
                              hipStream_t stream) {
  (void)in_sizes; (void)n_in; (void)out_size; (void)ws_size;
  const float* q    = (const float*)d_in[0];
  const float* k    = (const float*)d_in[1];
  const float* v    = (const float*)d_in[2];
  const float* beta = (const float*)d_in[3];

  const size_t NELEM = (size_t)NBH * LSEQ * DKD;   // 8,388,608

  bf16* gq = (bf16*)d_ws;
  bf16* gk = gq + NELEM;
  bf16* gw = gk + NELEM;
  float* gu = (float*)(gw + NELEM);

  float* o_out = (float*)d_out;
  float* S_out = o_out + NELEM;

  delta_prep<<<NBH * NCH, 32, 0, stream>>>(q, k, v, beta, gq, gk, gw, gu);

  const size_t smem = (128 * 128) * sizeof(float)      // S
                    + (128 * LDP) * sizeof(bf16)       // S^T bf16
                    + (128 * LDC) * sizeof(bf16)       // u_new^T
                    + 3 * (CSZ * LDP) * sizeof(bf16)   // q,k,w
                    + (CSZ * LDC) * sizeof(bf16);      // att
  delta_scan<<<NBH, 256, smem, stream>>>(gq, gk, gw, gu, o_out, S_out);
}
